// GradientSculptedAttention_74577812127904
// MI455X (gfx1250) — matrix operand order repaired and checked by reading, not compiled, not run on hardware
//
#include <hip/hip_runtime.h>
#include <hip/hip_bf16.h>

typedef __attribute__((ext_vector_type(16))) _Float16 v16h;
typedef __attribute__((ext_vector_type(8)))  _Float16 v8h;
typedef __attribute__((ext_vector_type(8)))  float    v8f;
typedef __attribute__((ext_vector_type(4)))  float    v4f;

#define HEADS    8
#define DIM_HEAD 64
#define BATCH    4
#define SEQ      2048
#define DMODEL   512
#define INNER    512
#define ROWS     (BATCH*SEQ)   /* 8192 */
#define NC3      (3*INNER)     /* 1536 */
#define NBINS    8192

// ---------- WMMA fragment helpers (layouts per CDNA5 ISA 7.12.2) ----------
__device__ __forceinline__ v8h ld8h(const _Float16* p) { return *(const v8h*)p; }
__device__ __forceinline__ v16h cat8(v8h lo, v8h hi) {
  return __builtin_shufflevector(lo, hi, 0,1,2,3,4,5,6,7,8,9,10,11,12,13,14,15);
}
// A fragment (16x32 f16): lane holds row (lane&15); halves 0..7 -> k = kb+8*hh..,
// halves 8..15 -> k = kb+16+8*hh..  (hh = lane>>4)
__device__ __forceinline__ v16h ldA(const _Float16* row, int kb, int hh) {
  return cat8(ld8h(row + kb + 8*hh), ld8h(row + kb + 16 + 8*hh));
}
// B fragment (32x16 f16) gathered from B^T stored n-major: lane holds col (lane&15);
// halves 0..7 -> k = kb+8*hh.., halves 8..15 -> k = kb+16+8*hh..  (same as A)
__device__ __forceinline__ v16h ldB(const _Float16* btrow, int kb, int hh) {
  return cat8(ld8h(btrow + kb + 8*hh), ld8h(btrow + kb + 16 + 8*hh));
}
__device__ __forceinline__ v8f wmma_f16(v16h a, v16h b, v8f c) {
  return __builtin_amdgcn_wmma_f32_16x16x32_f16(false, a, false, b, (short)0, c, false, false);
}

// ---------- conversion / prep kernels ----------
__global__ void cvt_f32_to_f16(const float* __restrict__ in, _Float16* __restrict__ out, int n) {
  int i = blockIdx.x*blockDim.x + threadIdx.x;
  if (i < n) out[i] = (_Float16)in[i];
}
// out[c][r] = in[r][c]; out is [cols_in][rows_in]
__global__ void cvt_transpose_f16(const float* __restrict__ in, _Float16* __restrict__ out,
                                  int rows_in, int cols_in) {
  int i = blockIdx.x*blockDim.x + threadIdx.x;
  if (i < rows_in*cols_in) {
    int c = i / rows_in, r = i % rows_in;
    out[i] = (_Float16)in[(size_t)r*cols_in + c];
  }
}
__global__ void zero_u32(unsigned* __restrict__ p, int n) {
  int i = blockIdx.x*blockDim.x + threadIdx.x;
  if (i < n) p[i] = 0u;
}

// ---------- QKV projection GEMM: [8192,512] x [512,1536] -> q/k/v f16 ----------
__global__ __launch_bounds__(32)
void qkv_gemm(const _Float16* __restrict__ xh, const _Float16* __restrict__ wqT,
              _Float16* __restrict__ qh, _Float16* __restrict__ kh, _Float16* __restrict__ vh) {
  int lane = threadIdx.x & 31;
  int l15 = lane & 15, hh = lane >> 4;
  int tn = blockIdx.x;      // 0..95
  int tm = blockIdx.y;      // 0..511
  const _Float16* arow = xh  + (size_t)(tm*16 + l15)*DMODEL;
  const _Float16* brow = wqT + (size_t)(tn*16 + l15)*DMODEL;
  v8f c = {};
#pragma unroll
  for (int k0 = 0; k0 < DMODEL; k0 += 32)
    c = wmma_f16(ldA(arow, k0, hh), ldB(brow, k0, hh), c);

  int col = tn*16 + l15;           // this lane's output column (fixed for all 8 rows)
  int which = col / INNER;         // 0=q 1=k 2=v
  int ic = col % INNER;
  int head = ic >> 6, dc = ic & 63;
  _Float16* dst = (which == 0) ? qh : (which == 1) ? kh : vh;
  float sc = (which == 0) ? 0.125f : 1.0f;   // fold 1/sqrt(64) into Q
#pragma unroll
  for (int r = 0; r < 8; ++r) {
    int gr = tm*16 + r + 8*hh;
    int b = gr >> 11, rowin = gr & 2047;
    size_t idx = (((size_t)b*HEADS + head)*SEQ + rowin)*DIM_HEAD + dc;
    dst[idx] = (_Float16)(c[r] * sc);
  }
}

// ---------- global histogram of grad_ema over [0,1) ----------
__global__ __launch_bounds__(256)
void hist_kernel(const float* __restrict__ ge, unsigned long long n,
                 unsigned* __restrict__ gbins) {
  __shared__ unsigned bins[NBINS];
  for (int i = threadIdx.x; i < NBINS; i += 256) bins[i] = 0u;
  __syncthreads();
  unsigned long long nv = n >> 2;
  unsigned long long stride = (unsigned long long)gridDim.x * 256ull;
  for (unsigned long long i = (unsigned long long)blockIdx.x*256ull + threadIdx.x;
       i < nv; i += stride) {
    v4f v = *(const v4f*)(ge + 4ull*i);
#pragma unroll
    for (int j = 0; j < 4; ++j) {
      int bx = (int)(v[j] * (float)NBINS);
      bx = bx < 0 ? 0 : (bx >= NBINS ? NBINS-1 : bx);
      atomicAdd(&bins[bx], 1u);
    }
  }
  if (blockIdx.x == 0 && threadIdx.x == 0) {    // tail (n not multiple of 4)
    for (unsigned long long i = nv << 2; i < n; ++i) {
      int bx = (int)(ge[i] * (float)NBINS);
      bx = bx < 0 ? 0 : (bx >= NBINS ? NBINS-1 : bx);
      atomicAdd(&bins[bx], 1u);
    }
  }
  __syncthreads();
  for (int i = threadIdx.x; i < NBINS; i += 256)
    if (bins[i]) atomicAdd(&gbins[i], bins[i]);
}

// ---------- 0.9 quantile from histogram (linear interpolation in-bin) ----------
__global__ void thresh_kernel(const unsigned* __restrict__ gbins, float* __restrict__ thr,
                              unsigned long long n) {
  if (threadIdx.x == 0 && blockIdx.x == 0) {
    double rank = 0.9 * ((double)n - 1.0);
    double cum = 0.0;
    float t = 1.0f;
    for (int i = 0; i < NBINS; ++i) {
      double c = (double)gbins[i];
      if (cum + c > rank) {
        double denom = c > 0.0 ? c : 1.0;
        t = (float)(((double)i + (rank - cum)/denom) / (double)NBINS);
        break;
      }
      cum += c;
    }
    thr[0] = t;
  }
}

// ---------- fused masked flash attention ----------
// grid: (SEQ/64, BATCH*HEADS), block: 128 (4 waves; each wave owns 16 q-rows)
__global__ __launch_bounds__(128)
void flash_attn(const _Float16* __restrict__ qh, const _Float16* __restrict__ kh,
                const _Float16* __restrict__ vh, const float* __restrict__ ge,
                const float* __restrict__ thrp, _Float16* __restrict__ attnh) {
  __shared__ __attribute__((aligned(16))) _Float16 Vt[DIM_HEAD][40];   // V^T tile (padded)
  __shared__ __attribute__((aligned(16))) _Float16 Pst[4][16][40];     // per-wave P staging

  int bh = blockIdx.y;                 // 0..31
  int b = bh >> 3, h = bh & 7;
  int wave = threadIdx.x >> 5, lane = threadIdx.x & 31;
  int l15 = lane & 15, hh = lane >> 4;
  int qrow0 = (blockIdx.x*4 + wave)*16;
  float thr = thrp[0];
  const float NEG_INF = -__builtin_inff();

  const _Float16* qb = qh + (size_t)bh*SEQ*DIM_HEAD;
  const _Float16* kb = kh + (size_t)bh*SEQ*DIM_HEAD;
  const _Float16* vb = vh + (size_t)bh*SEQ*DIM_HEAD;
  const float*    gb = ge + (size_t)bh*SEQ*SEQ;

  const _Float16* qrow = qb + (size_t)(qrow0 + l15)*DIM_HEAD;
  v16h qa0 = ldA(qrow, 0,  hh);        // dh 0..31
  v16h qa1 = ldA(qrow, 32, hh);        // dh 32..63

  float mrun[8], lrun[8];
  v8f acc0 = {}, acc1 = {}, acc2 = {}, acc3 = {};
#pragma unroll
  for (int r = 0; r < 8; ++r) { mrun[r] = NEG_INF; lrun[r] = 0.0f; }

  for (int jb = 0; jb < SEQ/32; ++jb) {
    int j0 = jb*32;
    __syncthreads();                   // previous iteration done with Vt
    // cooperative transposed V stage: Vt[dh][j]
    for (int idx = threadIdx.x; idx < 32*DIM_HEAD; idx += 128) {
      int r = idx >> 6, cc = idx & 63;
      Vt[cc][r] = vb[(size_t)(j0 + r)*DIM_HEAD + cc];
    }
    __syncthreads();

    // S = Q K^T  (two 16x16 col-tiles, dh accumulated over two K=32 WMMAs)
    const _Float16* kr0 = kb + (size_t)(j0 + l15)*DIM_HEAD;
    const _Float16* kr1 = kr0 + 16*DIM_HEAD;
    v8f s0 = {}, s1 = {};
    s0 = wmma_f16(qa0, ldB(kr0, 0,  hh), s0);
    s0 = wmma_f16(qa1, ldB(kr0, 32, hh), s0);
    s1 = wmma_f16(qa0, ldB(kr1, 0,  hh), s1);
    s1 = wmma_f16(qa1, ldB(kr1, 32, hh), s1);

    if (j0 + 32 < SEQ) __builtin_prefetch(vb + (size_t)(j0 + 32)*DIM_HEAD, 0, 1);

    // mask with grad_ema threshold + online softmax (rows live in 16-lane halves)
#pragma unroll
    for (int r = 0; r < 8; ++r) {
      int rr = r + 8*hh;
      const float* gp = gb + (size_t)(qrow0 + rr)*SEQ + j0;
      float g0 = gp[l15], g1 = gp[16 + l15];
      float v0 = (g0 >= thr) ? s0[r] : NEG_INF;
      float v1 = (g1 >= thr) ? s1[r] : NEG_INF;
      float mloc = fmaxf(v0, v1);
      mloc = fmaxf(mloc, __shfl_xor(mloc, 1, 32));
      mloc = fmaxf(mloc, __shfl_xor(mloc, 2, 32));
      mloc = fmaxf(mloc, __shfl_xor(mloc, 4, 32));
      mloc = fmaxf(mloc, __shfl_xor(mloc, 8, 32));
      float mnew = fmaxf(mrun[r], mloc);
      bool fin = (mnew > NEG_INF);
      float sc_old = fin ? __expf(mrun[r] - mnew) : 1.0f;
      float p0 = fin ? __expf(v0 - mnew) : 0.0f;
      float p1 = fin ? __expf(v1 - mnew) : 0.0f;
      float ps = p0 + p1;
      ps += __shfl_xor(ps, 1, 32);
      ps += __shfl_xor(ps, 2, 32);
      ps += __shfl_xor(ps, 4, 32);
      ps += __shfl_xor(ps, 8, 32);
      lrun[r] = lrun[r]*sc_old + ps;
      mrun[r] = mnew;
      acc0[r] *= sc_old; acc1[r] *= sc_old; acc2[r] *= sc_old; acc3[r] *= sc_old;
      Pst[wave][rr][l15]      = (_Float16)p0;
      Pst[wave][rr][16 + l15] = (_Float16)p1;
    }
    __syncthreads();                   // Pst (C-layout) -> A-layout via LDS

    // O += P V : P is 16x32 A-fragment, V^T rows give B-fragments for 4 dh tiles
    v16h pa = ldA(&Pst[wave][l15][0], 0, hh);
    acc0 = wmma_f16(pa, ldB(&Vt[ 0 + l15][0], 0, hh), acc0);
    acc1 = wmma_f16(pa, ldB(&Vt[16 + l15][0], 0, hh), acc1);
    acc2 = wmma_f16(pa, ldB(&Vt[32 + l15][0], 0, hh), acc2);
    acc3 = wmma_f16(pa, ldB(&Vt[48 + l15][0], 0, hh), acc3);
  }

  // epilogue: divide by l, write [b, n, h*dh] as f16 for the final GEMM
#pragma unroll
  for (int r = 0; r < 8; ++r) {
    int rr = r + 8*hh;
    float inv = lrun[r] > 0.0f ? 1.0f/lrun[r] : 0.0f;
    size_t row = (size_t)b*SEQ + qrow0 + rr;
    _Float16* op = attnh + row*INNER + h*DIM_HEAD + l15;
    op[0]  = (_Float16)(acc0[r]*inv);
    op[16] = (_Float16)(acc1[r]*inv);
    op[32] = (_Float16)(acc2[r]*inv);
    op[48] = (_Float16)(acc3[r]*inv);
  }
}

// ---------- output projection: [8192,512] x [512,512] + bias -> fp32 ----------
__global__ __launch_bounds__(32)
void out_gemm(const _Float16* __restrict__ attnh, const _Float16* __restrict__ woT,
              const float* __restrict__ bout, float* __restrict__ out) {
  int lane = threadIdx.x & 31;
  int l15 = lane & 15, hh = lane >> 4;
  int tn = blockIdx.x;   // 0..31
  int tm = blockIdx.y;   // 0..511
  const _Float16* arow = attnh + (size_t)(tm*16 + l15)*INNER;
  const _Float16* brow = woT   + (size_t)(tn*16 + l15)*INNER;
  v8f c = {};
#pragma unroll
  for (int k0 = 0; k0 < INNER; k0 += 32)
    c = wmma_f16(ldA(arow, k0, hh), ldB(brow, k0, hh), c);
  int col = tn*16 + l15;
  float bo = bout[col];
#pragma unroll
  for (int r = 0; r < 8; ++r) {
    int gr = tm*16 + r + 8*hh;
    out[(size_t)gr*DMODEL + col] = c[r] + bo;
  }
}

extern "C" void kernel_launch(void* const* d_in, const int* in_sizes, int n_in,
                              void* d_out, int out_size, void* d_ws, size_t ws_size,
                              hipStream_t stream) {
  (void)n_in; (void)out_size; (void)ws_size;
  const float* x     = (const float*)d_in[0];
  const float* ge    = (const float*)d_in[1];
  const float* w_qkv = (const float*)d_in[2];
  const float* w_out = (const float*)d_in[3];
  const float* b_out = (const float*)d_in[4];
  float* out = (float*)d_out;

  char* ws = (char*)d_ws;
  size_t off = 0;
  auto take = [&](size_t bytes) -> void* {
    void* p = ws + off;
    off = (off + bytes + 255) & ~(size_t)255;
    return p;
  };
  _Float16* xh    = (_Float16*)take((size_t)ROWS*DMODEL*2);                   // x in f16
  _Float16* wqT   = (_Float16*)take((size_t)NC3*DMODEL*2);                    // w_qkv^T f16
  _Float16* woT   = (_Float16*)take((size_t)DMODEL*INNER*2);                  // w_out^T f16
  _Float16* qh    = (_Float16*)take((size_t)BATCH*HEADS*SEQ*DIM_HEAD*2);
  _Float16* kh    = (_Float16*)take((size_t)BATCH*HEADS*SEQ*DIM_HEAD*2);
  _Float16* vh    = (_Float16*)take((size_t)BATCH*HEADS*SEQ*DIM_HEAD*2);
  _Float16* attnh = (_Float16*)take((size_t)ROWS*INNER*2);
  unsigned* bins  = (unsigned*)take((size_t)NBINS*4);
  float*    thr   = (float*)take(256);

  unsigned long long nge = (unsigned long long)in_sizes[1];

  int nx = ROWS*DMODEL;
  cvt_f32_to_f16<<<(nx + 255)/256, 256, 0, stream>>>(x, xh, nx);
  cvt_transpose_f16<<<(NC3*DMODEL + 255)/256, 256, 0, stream>>>(w_qkv, wqT, DMODEL, NC3);
  cvt_transpose_f16<<<(DMODEL*INNER + 255)/256, 256, 0, stream>>>(w_out, woT, DMODEL, INNER);
  zero_u32<<<(NBINS + 255)/256, 256, 0, stream>>>(bins, NBINS);
  hist_kernel<<<2048, 256, 0, stream>>>(ge, nge, bins);
  thresh_kernel<<<1, 32, 0, stream>>>(bins, thr, nge);
  qkv_gemm<<<dim3(NC3/16, ROWS/16), 32, 0, stream>>>(xh, wqT, qh, kh, vh);
  flash_attn<<<dim3(SEQ/64, BATCH*HEADS), 128, 0, stream>>>(qh, kh, vh, ge, thr, attnh);
  out_gemm<<<dim3(DMODEL/16, ROWS/16), 32, 0, stream>>>(attnh, woT, b_out, out);
}
